// YaAttention_74225624809490
// MI455X (gfx1250) — compile-verified, matrix operand order checked by reading, not run on hardware
//
#include <hip/hip_runtime.h>

// Problem constants (B, S, E, H, DH, W) = (2, 4096, 1024, 16, 64, 512)
#define BB   2
#define SS   4096
#define EE   1024
#define HH   16
#define DHH  64
#define WW   512
#define NBB  8          // S / W
#define HD   1024       // H*DH

typedef __attribute__((ext_vector_type(16))) __bf16 v16bf;
typedef __attribute__((ext_vector_type(8)))  float  v8f;

__device__ __forceinline__ v8f wmma_bf16(v16bf a, v16bf b, v8f c) {
  // D = A(16x32 bf16) * B(32x16 bf16) + C(16x16 f32)
  return __builtin_amdgcn_wmma_f32_16x16x32_bf16(
      /*neg_a=*/false, a, /*neg_b=*/false, b,
      /*c_mod=*/(short)0, c, /*reuse_a=*/false, /*reuse_b=*/false);
}

// Load a 16x32 A-frag or 32x16 B-frag per the CDNA5 16-bit layout:
// lane holds 8 contiguous elems at +0 and 8 contiguous elems at +16
// (caller bakes in row base + kb = (lane<16 ? 0 : 8)).
__device__ __forceinline__ v16bf load_frag(const __bf16* p) {
  union { uint4 u[2]; v16bf v; } r;
  r.u[0] = *reinterpret_cast<const uint4*>(p);
  r.u[1] = *reinterpret_cast<const uint4*>(p + 16);
  return r.v;
}

__device__ __forceinline__ v8f zero8() {
  v8f z;
#pragma unroll
  for (int r = 0; r < 8; ++r) z[r] = 0.0f;
  return z;
}

__device__ __forceinline__ void wait_ds0() {
#if __has_builtin(__builtin_amdgcn_s_wait_dscnt)
  __builtin_amdgcn_s_wait_dscnt(0);
#else
  asm volatile("s_wait_dscnt 0" ::: "memory");
#endif
}

// ---------------------------------------------------------------- converters

__global__ __launch_bounds__(256) void cast_bf16_kernel(
    const float* __restrict__ in, __bf16* __restrict__ out, int n) {
  int i = blockIdx.x * 256 + threadIdx.x;
  if (i < n) out[i] = (__bf16)in[i];
}

// out[r*C + c] = in[c*R + r]  (transpose + cast), total R*C elements
__global__ __launch_bounds__(256) void transpose_bf16_kernel(
    const float* __restrict__ in, __bf16* __restrict__ out, int R, int C) {
  int idx = blockIdx.x * 256 + threadIdx.x;
  if (idx >= R * C) return;
  int r = idx / C;
  int c = idx - r * C;
  out[idx] = (__bf16)in[(size_t)c * R + r];
}

// wkvt[n][e] : n<64 -> Wk[e][n], n>=64 -> Wv[e][n-64]; shape [128][1024]
__global__ __launch_bounds__(256) void build_wkv_kernel(
    const float* __restrict__ Wk, const float* __restrict__ Wv,
    __bf16* __restrict__ out) {
  int idx = blockIdx.x * 256 + threadIdx.x;   // 128*1024 total
  int n = idx >> 10;
  int e = idx & 1023;
  float v = (n < 64) ? Wk[(size_t)e * 64 + n] : Wv[(size_t)e * 64 + (n - 64)];
  out[idx] = (__bf16)v;
}

// ---------------------------------------------------------------- bf16 GEMM
// C[M,N] = A[M,K] * Bt[N,K]^T ; each wave computes a 32x64 tile.
// 256 threads = 8 waves per block.

template <bool OUT_BF16>
__global__ __launch_bounds__(256) void gemm_bf16_kernel(
    const __bf16* __restrict__ A, const __bf16* __restrict__ Bt,
    float* __restrict__ Cf, __bf16* __restrict__ Ch, int M, int N, int K) {
  const int tid  = threadIdx.x;
  const int lane = tid & 31;
  const int wv   = tid >> 5;
  const int wid  = blockIdx.x * 8 + wv;
  const int tilesN = N >> 6;
  const int tm = wid / tilesN;
  const int tn = wid - tm * tilesN;
  const int m0 = tm * 32;
  const int n0 = tn * 64;
  const int ln = lane & 15;
  const int kb = (lane < 16) ? 0 : 8;

  v8f c[2][4];
#pragma unroll
  for (int mi = 0; mi < 2; ++mi)
#pragma unroll
    for (int f = 0; f < 4; ++f) c[mi][f] = zero8();

  const __bf16* arow0 = A + (size_t)(m0 + ln) * K + kb;
  const __bf16* arow1 = arow0 + (size_t)16 * K;
  const __bf16* brow[4];
#pragma unroll
  for (int f = 0; f < 4; ++f)
    brow[f] = Bt + (size_t)(n0 + f * 16 + ln) * K + kb;

  for (int k0 = 0; k0 < K; k0 += 32) {
    __builtin_prefetch(arow0 + k0 + 128, 0, 0);   // global_prefetch_b8
    v16bf a0 = load_frag(arow0 + k0);
    v16bf a1 = load_frag(arow1 + k0);
#pragma unroll
    for (int f = 0; f < 4; ++f) {
      v16bf bf = load_frag(brow[f] + k0);
      c[0][f] = wmma_bf16(a0, bf, c[0][f]);
      c[1][f] = wmma_bf16(a1, bf, c[1][f]);
    }
  }

  // C/D layout: VGPR r -> row (lane<16 ? r : r+8), col = lane&15
#pragma unroll
  for (int mi = 0; mi < 2; ++mi) {
#pragma unroll
    for (int f = 0; f < 4; ++f) {
#pragma unroll
      for (int r = 0; r < 8; ++r) {
        int row = m0 + mi * 16 + ((lane < 16) ? r : r + 8);
        int col = n0 + f * 16 + ln;
        if (OUT_BF16)
          Ch[(size_t)row * N + col] = (__bf16)c[mi][f][r];
        else
          Cf[(size_t)row * N + col] = c[mi][f][r];
      }
    }
  }
}

// ------------------------------------------------------- flash attention
// grid = B*H*NB*4 blocks of 256 threads (8 waves); each block: 128 queries;
// each wave: 16 query rows, online softmax over 8 chunks of 128 keys.

__global__ __launch_bounds__(256) void attn_kernel(
    const __bf16* __restrict__ qb, const __bf16* __restrict__ kvb,
    const float* __restrict__ bias, __bf16* __restrict__ attnb) {
  __shared__ __align__(16) __bf16 Klds[128][72];    // keys x dh (+pad)
  __shared__ __align__(16) __bf16 Vtlds[64][136];   // dh x keys (+pad)
  __shared__ __align__(16) __bf16 Plds[8][16][40];  // per-wave P staging
  __shared__ float biaslds[WW];

  const int tid  = threadIdx.x;
  const int lane = tid & 31;
  const int wv   = tid >> 5;
  const int ln   = lane & 15;
  const int hiL  = lane >> 4;                 // 0 or 1
  const int kb   = hiL ? 8 : 0;

  const int wg      = blockIdx.x;
  const int quarter = wg & 3;
  const int n       = (wg >> 2) & 7;
  const int h       = (wg >> 5) & 15;
  const int b       = wg >> 9;

  const int kstart = n * WW - WW;             // first key of the 2W window
  const int q0     = n * WW + quarter * 128 + wv * 16;
  const size_t bS  = (size_t)b * SS;
  const float scale = 0.125f;                 // 1/sqrt(64)

  for (int i = tid; i < WW; i += 256) biaslds[i] = bias[h * WW + i];

  // Q fragments (16 rows x 64 dh = two 16x32 A-frags)
  v16bf aq[2];
  {
    const __bf16* qp = qb + (bS + q0 + ln) * HD + h * DHH + kb;
    aq[0] = load_frag(qp);
    aq[1] = load_frag(qp + 32);
  }

  float m[8], l[8];
  v8f o[4];
#pragma unroll
  for (int r = 0; r < 8; ++r) { m[r] = -1e30f; l[r] = 0.0f; }
#pragma unroll
  for (int f = 0; f < 4; ++f) o[f] = zero8();

  for (int c = 0; c < 8; ++c) {
    __syncthreads();
    // ---- cooperative K/V chunk load: 128 keys x 64 dh each
    {
      const int u    = tid >> 1;
      const int half = tid & 1;
      const int j    = kstart + c * 128 + u;
      union { uint4 u4[4]; __bf16 hh[32]; } kk, vvv;
      if (j >= 0) {
        const uint4* s4 = reinterpret_cast<const uint4*>(kvb + (bS + j) * 128);
#pragma unroll
        for (int q = 0; q < 4; ++q) {
          kk.u4[q]  = s4[half * 4 + q];
          vvv.u4[q] = s4[8 + half * 4 + q];
        }
      } else {
        uint4 z; z.x = z.y = z.z = z.w = 0u;
#pragma unroll
        for (int q = 0; q < 4; ++q) { kk.u4[q] = z; vvv.u4[q] = z; }
      }
      uint4* kd = reinterpret_cast<uint4*>(&Klds[u][half * 32]);
#pragma unroll
      for (int q = 0; q < 4; ++q) kd[q] = kk.u4[q];
#pragma unroll
      for (int d2 = 0; d2 < 32; ++d2) Vtlds[half * 32 + d2][u] = vvv.hh[d2];
    }
    __syncthreads();

    // ---- scores S = Q * K^T for 8 N-tiles of 16 keys
    v8f st[8];
#pragma unroll
    for (int t = 0; t < 8; ++t) {
      const __bf16* kp = &Klds[t * 16 + ln][kb];
      v16bf b0 = load_frag(kp);
      v16bf b1 = load_frag(kp + 32);
      v8f acc = zero8();
      acc = wmma_bf16(aq[0], b0, acc);
      acc = wmma_bf16(aq[1], b1, acc);
      st[t] = acc;
    }

    // ---- scale + relative bias + causal sliding-window mask
    const int jbase = kstart + c * 128 + ln;
#pragma unroll
    for (int t = 0; t < 8; ++t) {
      const int j = jbase + t * 16;
#pragma unroll
      for (int r = 0; r < 8; ++r) {
        const int i = q0 + r + hiL * 8;
        const int d = i - j;
        const bool valid = (j >= 0) & (d >= 0) & (d < WW);
        const int dc = (d < 0) ? 0 : ((d >= WW) ? (WW - 1) : d);
        const float sv = st[t][r] * scale + biaslds[dc];
        st[t][r] = valid ? sv : -1e30f;
      }
    }

    // ---- online softmax: row max across tiles + 16 lanes
    float rmax[8];
#pragma unroll
    for (int r = 0; r < 8; ++r) {
      float v = st[0][r];
#pragma unroll
      for (int t = 1; t < 8; ++t) v = fmaxf(v, st[t][r]);
      rmax[r] = v;
    }
#pragma unroll
    for (int r = 0; r < 8; ++r) {
#pragma unroll
      for (int off = 1; off < 16; off <<= 1)
        rmax[r] = fmaxf(rmax[r], __shfl_xor(rmax[r], off, 32));
    }
    float mnew[8], corr[8], rsum[8];
#pragma unroll
    for (int r = 0; r < 8; ++r) {
      mnew[r] = fmaxf(m[r], rmax[r]);
      corr[r] = __expf(m[r] - mnew[r]);
      l[r] *= corr[r];
      rsum[r] = 0.0f;
      m[r] = mnew[r];
    }
#pragma unroll
    for (int f = 0; f < 4; ++f)
#pragma unroll
      for (int r = 0; r < 8; ++r) o[f][r] *= corr[r];

    // ---- P = exp(S - mnew); stage to LDS (C->A layout), then P*V
#pragma unroll
    for (int kk2 = 0; kk2 < 4; ++kk2) {
#pragma unroll
      for (int tt = 0; tt < 2; ++tt) {
        const int t = kk2 * 2 + tt;
#pragma unroll
        for (int r = 0; r < 8; ++r) {
          float p = (mnew[r] < -0.9e30f) ? 0.0f : __expf(st[t][r] - mnew[r]);
          rsum[r] += p;
          Plds[wv][r + hiL * 8][tt * 16 + ln] = (__bf16)p;
        }
      }
      wait_ds0();   // per-wave LDS ordering before A-frag readback
      v16bf ap = load_frag(&Plds[wv][ln][kb]);
#pragma unroll
      for (int f = 0; f < 4; ++f) {
        v16bf bv = load_frag(&Vtlds[f * 16 + ln][kk2 * 32 + kb]);
        o[f] = wmma_bf16(ap, bv, o[f]);
      }
    }

    // ---- fold row sums into l
#pragma unroll
    for (int r = 0; r < 8; ++r) {
#pragma unroll
      for (int off = 1; off < 16; off <<= 1)
        rsum[r] += __shfl_xor(rsum[r], off, 32);
      l[r] += rsum[r];
    }
  }

  // ---- epilogue: normalize and write attn (bf16)
#pragma unroll
  for (int r = 0; r < 8; ++r) {
    const float inv = 1.0f / l[r];
    const int i = q0 + r + hiL * 8;
    __bf16* dst = attnb + (bS + i) * HD + h * DHH + ln;
#pragma unroll
    for (int f = 0; f < 4; ++f) dst[f * 16] = (__bf16)(o[f][r] * inv);
  }
}

// ---------------------------------------------------------------- launcher

extern "C" void kernel_launch(void* const* d_in, const int* in_sizes, int n_in,
                              void* d_out, int out_size, void* d_ws, size_t ws_size,
                              hipStream_t stream) {
  (void)in_sizes; (void)n_in; (void)out_size; (void)ws_size;
  const float* x    = (const float*)d_in[0];
  const float* bias = (const float*)d_in[1];
  const float* Wq   = (const float*)d_in[2];
  const float* Wk   = (const float*)d_in[3];
  const float* Wv   = (const float*)d_in[4];
  const float* Wo   = (const float*)d_in[5];
  float* out = (float*)d_out;

  char* ws = (char*)d_ws;
  __bf16* xb    = (__bf16*)(ws + 0);           // [8192 x 1024]  16 MB
  __bf16* qbuf  = (__bf16*)(ws + 16777216);    // [8192 x 1024]  16 MB
  __bf16* kvb   = (__bf16*)(ws + 33554432);    // [8192 x 128]    2 MB
  __bf16* attnb = (__bf16*)(ws + 35651584);    // [8192 x 1024]  16 MB
  __bf16* wqt   = (__bf16*)(ws + 52428800);    // [1024 x 1024]   2 MB
  __bf16* wkvt  = (__bf16*)(ws + 54525952);    // [128 x 1024]  256 KB
  __bf16* wot   = (__bf16*)(ws + 54788096);    // [1024 x 1024]   2 MB

  const int BSE = BB * SS * EE;                // 8,388,608
  cast_bf16_kernel<<<(BSE + 255) / 256, 256, 0, stream>>>(x, xb, BSE);
  transpose_bf16_kernel<<<(1024 * 1024) / 256, 256, 0, stream>>>(Wq, wqt, 1024, 1024);
  build_wkv_kernel<<<(128 * 1024) / 256, 256, 0, stream>>>(Wk, Wv, wkvt);
  transpose_bf16_kernel<<<(1024 * 1024) / 256, 256, 0, stream>>>(Wo, wot, 1024, 1024);

  // Q = x * Wq   : [8192,1024] x [1024,1024] -> bf16
  gemm_bf16_kernel<true><<<512, 256, 0, stream>>>(xb, wqt, nullptr, qbuf,
                                                  8192, 1024, 1024);
  // KV = x * [Wk|Wv] : [8192,1024] x [1024,128] -> bf16
  gemm_bf16_kernel<true><<<64, 256, 0, stream>>>(xb, wkvt, nullptr, kvb,
                                                 8192, 128, 1024);
  // sliding-window attention
  attn_kernel<<<BB * HH * NBB * 4, 256, 0, stream>>>(qbuf, kvb, bias, attnb);
  // out = attn * WoFlat : [8192,1024] x [1024,1024] -> f32
  gemm_bf16_kernel<false><<<512, 256, 0, stream>>>(attnb, wot, out, nullptr,
                                                   8192, 1024, 1024);
}
